// Attention_16913581212014
// MI455X (gfx1250) — matrix operand order repaired and checked by reading, not compiled, not run on hardware
//
#include <hip/hip_runtime.h>
#include <hip/hip_bf16.h>

// ---------------------------------------------------------------------------
// Problem constants (match the reference)
// ---------------------------------------------------------------------------
constexpr int Bq   = 32;
constexpr int Nq   = 1024;
constexpr int DIN  = 512;
constexpr int DOUT = 512;
constexpr int Hq   = 8;
constexpr int DH   = 64;          // head dim
constexpr int NT   = Nq / 16;     // 64 row tiles of 16

typedef __bf16 bf16_t;
typedef __attribute__((ext_vector_type(8)))  __bf16 v8bf;
typedef __attribute__((ext_vector_type(16))) __bf16 v16bf;
typedef __attribute__((ext_vector_type(8)))  float  v8f;

// round-to-nearest-even f32 -> bf16, pure integer ops (no libcalls)
__device__ inline bf16_t f2bf(float f) {
    unsigned u = __builtin_bit_cast(unsigned, f);
    u += 0x7FFFu + ((u >> 16) & 1u);
    unsigned short s = (unsigned short)(u >> 16);
    return __builtin_bit_cast(bf16_t, s);
}

__device__ inline v16bf cat16(v8bf a, v8bf b) {
    return __builtin_shufflevector(a, b, 0,1,2,3,4,5,6,7,8,9,10,11,12,13,14,15);
}
__device__ inline v8bf ld8(const bf16_t* p) { return *(const v8bf*)p; }

// D(16x16 f32) = A(16x32 bf16) * B(32x16 bf16) + C, f32 accumulate
__device__ inline v8f wmma_bf16(v16bf a, v16bf b, v8f c) {
    return __builtin_amdgcn_wmma_f32_16x16x32_bf16(
        /*neg_a=*/false, a, /*neg_b=*/false, b,
        /*c_mod=*/(short)0, c, /*reuse_a=*/false, /*reuse_b=*/false);
}

// CDNA5 async copy: memory -> LDS directly (no VGPR round trip), ASYNCcnt.
// Per-lane: LDS[vdst] = MEM[vaddr], 8 bytes.  (GV addressing mode)
__device__ inline void async_load_lds_b64(unsigned lds_off, const void* gptr) {
    unsigned long long ga = (unsigned long long)gptr;
    asm volatile("global_load_async_to_lds_b64 %0, %1, off"
                 :: "v"(lds_off), "v"(ga)
                 : "memory");
}
__device__ inline void wait_asynccnt0() {
    asm volatile("s_wait_asynccnt 0x0" ::: "memory");
}

// ---------------------------------------------------------------------------
// Kernel 0a: X f32 -> bf16
// ---------------------------------------------------------------------------
__global__ void cvt_x_kernel(const float* __restrict__ X,
                             bf16_t* __restrict__ X16, int n) {
    int i = blockIdx.x * blockDim.x + threadIdx.x;
    if (i < n) X16[i] = f2bf(X[i]);
}

// ---------------------------------------------------------------------------
// Kernel 0b: Wp [H,DIN,DH] -> WpT16 [H,DH,DIN] bf16 (so WMMA B-operand loads
// are contiguous b128s), C [H,DH,DH] -> bf16 (row-major already B-friendly)
// ---------------------------------------------------------------------------
__global__ void cvt_w_kernel(const float* __restrict__ Wp,
                             const float* __restrict__ C,
                             bf16_t* __restrict__ WpT16,
                             bf16_t* __restrict__ C16) {
    int i = blockIdx.x * blockDim.x + threadIdx.x;
    if (i < Hq * DIN * DH) {
        int h = i / (DIN * DH);
        int r = i % (DIN * DH);
        int d = r / DH;
        int k = r % DH;
        WpT16[((size_t)h * DH + k) * DIN + d] = f2bf(Wp[i]);
    }
    if (i < Hq * DH * DH) C16[i] = f2bf(C[i]);
}

// ---------------------------------------------------------------------------
// Kernel 1: per-head projection  Xp[b,h,n,k] = X[b,n,:] @ Wp[h,:,k] + bp[h,k]
// One wave per (b, ntile, h): 16x64 tile, K=512 -> 64 WMMAs. Fully-unrolled
// software pipeline: chunk k+32 operand loads issue before chunk k WMMAs, and
// SSA renaming absorbs the buffer rotation (no v_mov traffic).
// ---------------------------------------------------------------------------
__global__ __launch_bounds__(256)
void proj_kernel(const bf16_t* __restrict__ X16,
                 const bf16_t* __restrict__ WpT16,
                 const float*  __restrict__ bp,
                 bf16_t* __restrict__ Xp16,
                 bf16_t* __restrict__ XpT16) {
    const int b      = blockIdx.x >> 6;
    const int base_n = (blockIdx.x & 63) * 16;
    const int h      = threadIdx.x >> 5;          // wave = head
    const int lane   = threadIdx.x & 31;
    const int lm     = lane & 15;
    const bool hi    = lane >= 16;
    const int ash    = hi ? 16 : 0;               // A-operand K shift
    const int bsh    = hi ? 16 : 0;               // B-operand K shift

    v8f acc[4] = {};
    const bf16_t* xrow  = X16 + ((size_t)b * Nq + base_n + lm) * DIN;
    const bf16_t* wbase = WpT16 + (size_t)h * DH * DIN;

    // prologue: chunk 0 operands
    v16bf a_cur = cat16(ld8(xrow + ash), ld8(xrow + ash + 8));
    v16bf b_cur[4];
#pragma unroll
    for (int cb = 0; cb < 4; ++cb) {
        const bf16_t* w = wbase + (size_t)(cb * 16 + lm) * DIN + bsh;
        b_cur[cb] = cat16(ld8(w), ld8(w + 8));
    }

    // steady state (FULLY unrolled): issue chunk k+32 loads, then WMMA chunk k
#pragma unroll
    for (int ko = 0; ko < DIN - 32; ko += 32) {
        const int kn = ko + 32;
        v16bf a_nxt = cat16(ld8(xrow + kn + ash), ld8(xrow + kn + ash + 8));
        v16bf b_nxt[4];
#pragma unroll
        for (int cb = 0; cb < 4; ++cb) {
            const bf16_t* w = wbase + (size_t)(cb * 16 + lm) * DIN + kn + bsh;
            b_nxt[cb] = cat16(ld8(w), ld8(w + 8));
        }
#pragma unroll
        for (int cb = 0; cb < 4; ++cb)
            acc[cb] = wmma_bf16(a_cur, b_cur[cb], acc[cb]);
        a_cur = a_nxt;
#pragma unroll
        for (int cb = 0; cb < 4; ++cb) b_cur[cb] = b_nxt[cb];
    }
    // epilogue chunk
#pragma unroll
    for (int cb = 0; cb < 4; ++cb)
        acc[cb] = wmma_bf16(a_cur, b_cur[cb], acc[cb]);

    // bias + stores: Xp row-major (scalar b16) and XpT packed b128 per lane
#pragma unroll
    for (int cb = 0; cb < 4; ++cb) {
        const int col = cb * 16 + lm;
        const float bias = bp[h * DH + col];
        v8bf pk;
#pragma unroll
        for (int r = 0; r < 8; ++r) {
            const int row = base_n + r + (hi ? 8 : 0);
            bf16_t v = f2bf(acc[cb][r] + bias);
            Xp16[(((size_t)b * Hq + h) * Nq + row) * DH + col] = v;
            pk[r] = v;
        }
        // 8 consecutive rows of the [B,H,DH,N] layout -> one 16B store
        *(v8bf*)(XpT16 + (((size_t)b * Hq + h) * DH + col) * Nq
                       + base_n + (hi ? 8 : 0)) = pk;
    }
}

// ---------------------------------------------------------------------------
// Kernel 2: bilinear  Xc[b,h,n,j] = sum_k Xp[b,h,n,k] * C[h,j,k]
// One wave per (b, ntile, h): K=64 -> 8 WMMAs.
// ---------------------------------------------------------------------------
__global__ __launch_bounds__(256)
void bilinear_kernel(const bf16_t* __restrict__ Xp16,
                     const bf16_t* __restrict__ C16,
                     bf16_t* __restrict__ Xc16) {
    const int b      = blockIdx.x >> 6;
    const int base_n = (blockIdx.x & 63) * 16;
    const int h      = threadIdx.x >> 5;
    const int lane   = threadIdx.x & 31;
    const int lm     = lane & 15;
    const bool hi    = lane >= 16;

    v8f acc[4] = {};
    const bf16_t* xrow = Xp16 + (((size_t)b * Hq + h) * Nq + base_n + lm) * DH;

#pragma unroll
    for (int ko = 0; ko < DH; ko += 32) {
        const int ao = ko + (hi ? 16 : 0);
        v16bf avec = cat16(ld8(xrow + ao), ld8(xrow + ao + 8));
        const int bo = ko + (hi ? 16 : 0);
#pragma unroll
        for (int cb = 0; cb < 4; ++cb) {
            // B(K=k, N=j) = C[h][j][k]  -> row j of C is contiguous in k
            const bf16_t* crow = C16 + ((size_t)h * DH + cb * 16 + lm) * DH + bo;
            v16bf bvec = cat16(ld8(crow), ld8(crow + 8));
            acc[cb] = wmma_bf16(avec, bvec, acc[cb]);
        }
    }
#pragma unroll
    for (int cb = 0; cb < 4; ++cb) {
        const int col = cb * 16 + lm;
#pragma unroll
        for (int r = 0; r < 8; ++r) {
            const int row = base_n + r + (hi ? 8 : 0);
            Xc16[(((size_t)b * Hq + h) * Nq + row) * DH + col] = f2bf(acc[cb][r]);
        }
    }
}

// ---------------------------------------------------------------------------
// Kernel 3: fused masked attention + aggregation (flash-style, no [N,N] in HBM)
//   S = Xc_tile @ Xp^T ; attn = tanh(A * S) ; Out_tile += attn @ Xp
// Block = 8 waves = 8 heads sharing one (b, 16-row) tile; the adjacency mask
// tile is head-independent -> async-copied into LDS once, shared by all heads.
// attn tile is re-laid out D->A through LDS (the cross-lane transpose).
// ---------------------------------------------------------------------------
__global__ __launch_bounds__(256)
void attn_kernel(const float*  __restrict__ Amask,
                 const bf16_t* __restrict__ Xc16,
                 const bf16_t* __restrict__ Xp16,
                 const bf16_t* __restrict__ XpT16,
                 float* __restrict__ Out) {
    __shared__ __align__(16) float  maskT[16][32];        // 2 KB, shared by heads
    __shared__ __align__(16) bf16_t attnT[Hq][16][32];    // 1 KB per wave

    const int b      = blockIdx.x >> 6;
    const int base_n = (blockIdx.x & 63) * 16;
    const int h      = threadIdx.x >> 5;
    const int lane   = threadIdx.x & 31;
    const int lm     = lane & 15;
    const bool hi    = lane >= 16;

    // A-operand (resident all loop): Xc rows of this tile, K = DH = 64
    const bf16_t* xcrow =
        Xc16 + (((size_t)b * Hq + h) * Nq + base_n + lm) * DH;
    const int ash = hi ? 16 : 0;
    v16bf axc0 = cat16(ld8(xcrow + ash),      ld8(xcrow + ash + 8));
    v16bf axc1 = cat16(ld8(xcrow + 32 + ash), ld8(xcrow + 40 + ash));

    v8f out[4] = {};

    const float*  Arow = Amask + ((size_t)b * Nq + base_n) * Nq;
    const bf16_t* xph  = Xp16  + ((size_t)b * Hq + h) * (size_t)Nq * DH;
    const bf16_t* xpt  = XpT16 + ((size_t)b * Hq + h) * (size_t)DH * Nq;

    // per-thread mask-tile slot: 256 threads x 8 bytes = 16x32 f32 tile
    const int mrow = threadIdx.x >> 4;        // 0..15
    const int mcol = (threadIdx.x & 15) * 2;  // 0..30
    const unsigned mask_lds = (unsigned)(uintptr_t)&maskT[mrow][mcol];

    for (int mb = 0; mb < Nq; mb += 32) {
        __syncthreads();   // previous iteration finished reading maskT
        {   // async global->LDS copy of the mask tile (ASYNCcnt path)
            const float* mp = Arow + (size_t)mrow * Nq + mb + mcol;
            async_load_lds_b64(mask_lds, mp);
            if (mb + 32 < Nq)  // stream the next mask tile toward L2
                __builtin_prefetch(mp + 32, 0, 1);
            wait_asynccnt0();
        }
        __syncthreads();

        // ---- scores for two 16-column halves, mask * tanh, spill to LDS ----
#pragma unroll
        for (int half = 0; half < 2; ++half) {
            // B(K=j, N=m') = Xp[mb+16*half+m', j]: contiguous per lane row
            const bf16_t* brow = xph + (size_t)(mb + half * 16 + lm) * DH;
            const int j0 = hi ? 16 : 0;
            v16bf b0 = cat16(ld8(brow + j0),      ld8(brow + j0 + 8));
            v16bf b1 = cat16(ld8(brow + 32 + j0), ld8(brow + 32 + j0 + 8));
            v8f s = {};
            s = wmma_bf16(axc0, b0, s);
            s = wmma_bf16(axc1, b1, s);

            const int rbase = hi ? 8 : 0;
            const int ccol  = half * 16 + lm;
#pragma unroll
            for (int r = 0; r < 8; ++r) {
                float a = tanhf(maskT[rbase + r][ccol] * s[r]);
                attnT[h][rbase + r][ccol] = f2bf(a);   // ds_store_b16
            }
        }
        // LDS RAW fence before cross-lane re-read of the attn tile
        asm volatile("s_wait_dscnt 0x0" ::: "memory");

        // ---- reload attn tile in WMMA A-operand layout (16x32, K=m) ----
        const bf16_t* arow = &attnT[h][lm][0];
        const int o = hi ? 16 : 0;
        v16bf aat = cat16(*(const v8bf*)(arow + o),
                          *(const v8bf*)(arow + 8 + o));

        // ---- aggregation: Out += attn(16x32) @ Xp(32x64) ----
#pragma unroll
        for (int cb = 0; cb < 4; ++cb) {
            // B(K=mm, N=kk) = XpT[kb+kk, mb+mm]: contiguous per lane row
            const bf16_t* prow =
                xpt + (size_t)(cb * 16 + lm) * Nq + mb + (hi ? 16 : 0);
            v16bf bv = cat16(ld8(prow), ld8(prow + 8));
            out[cb] = wmma_bf16(aat, bv, out[cb]);
        }
    }

    // epilogue: relu + head-concat store  Out[b, n, h*DH + k]
#pragma unroll
    for (int cb = 0; cb < 4; ++cb) {
        const int col = h * DH + cb * 16 + lm;
#pragma unroll
        for (int r = 0; r < 8; ++r) {
            const int row = base_n + r + (hi ? 8 : 0);
            Out[((size_t)b * Nq + row) * DOUT + col] = fmaxf(out[cb][r], 0.0f);
        }
    }
}

// ---------------------------------------------------------------------------
// Launch
// ---------------------------------------------------------------------------
extern "C" void kernel_launch(void* const* d_in, const int* in_sizes, int n_in,
                              void* d_out, int out_size, void* d_ws, size_t ws_size,
                              hipStream_t stream) {
    const float* X  = (const float*)d_in[0];   // [B,N,DIN]
    const float* Am = (const float*)d_in[1];   // [B,N,N]
    const float* Wp = (const float*)d_in[2];   // [H,DIN,DH]
    const float* bp = (const float*)d_in[3];   // [H,DH]
    const float* C  = (const float*)d_in[4];   // [H,DH,DH]
    float* Out = (float*)d_out;                // [B,N,DOUT]

    // workspace carve-up (bf16 tensors), ~128.6 MB total
    char* ws = (char*)d_ws;
    const size_t SZ_X16  = (size_t)Bq * Nq * DIN * 2;       // 32 MB
    const size_t SZ_WPT  = (size_t)Hq * DH * DIN * 2;       // 512 KB
    const size_t SZ_C16  = (size_t)Hq * DH * DH * 2;        // 64 KB
    const size_t SZ_XP   = (size_t)Bq * Hq * Nq * DH * 2;   // 32 MB
    bf16_t* X16   = (bf16_t*)(ws);
    bf16_t* WpT16 = (bf16_t*)(ws + SZ_X16);
    bf16_t* C16   = (bf16_t*)(ws + SZ_X16 + SZ_WPT);
    bf16_t* Xp16  = (bf16_t*)(ws + SZ_X16 + SZ_WPT + SZ_C16);
    bf16_t* XpT16 = (bf16_t*)(ws + SZ_X16 + SZ_WPT + SZ_C16 + SZ_XP);
    bf16_t* Xc16  = (bf16_t*)(ws + SZ_X16 + SZ_WPT + SZ_C16 + 2 * SZ_XP);

    const int nx = Bq * Nq * DIN;
    cvt_x_kernel<<<(nx + 255) / 256, 256, 0, stream>>>(X, X16, nx);
    const int nw = Hq * DIN * DH;
    cvt_w_kernel<<<(nw + 255) / 256, 256, 0, stream>>>(Wp, C, WpT16, C16);

    dim3 grid(Bq * NT);   // 2048 blocks; 256 threads = 8 waves = 8 heads
    proj_kernel    <<<grid, 256, 0, stream>>>(X16, WpT16, bp, Xp16, XpT16);
    bilinear_kernel<<<grid, 256, 0, stream>>>(Xp16, C16, Xc16);
    attn_kernel    <<<grid, 256, 0, stream>>>(Am, Xc16, Xp16, XpT16, Out);
}
